// Sam3TrackerVideoRoPEAttention_73014444032502
// MI455X (gfx1250) — matrix operand order repaired and checked by reading, not compiled, not run on hardware
//
#include <hip/hip_runtime.h>

#define HIDDEN 256
#define SQ     4096
#define SK_TOT 16448
#define KT     32
#define NKT    (SK_TOT / KT)   // 514

typedef _Float16 h16;
typedef __attribute__((ext_vector_type(16))) _Float16 v16h;
typedef __attribute__((ext_vector_type(8)))  _Float16 v8h;
typedef __attribute__((ext_vector_type(8)))  float    v8f;
typedef __attribute__((ext_vector_type(4)))  float    v4f;
typedef __attribute__((ext_vector_type(4)))  unsigned int u32x4;
typedef __attribute__((ext_vector_type(8)))  unsigned int u32x8;

static __device__ __forceinline__ v8f wmma16(v16h a, v16h b, v8f c) {
  // D = A(16x32 f16) * B(32x16 f16) + C(16x16 f32)
  return __builtin_amdgcn_wmma_f32_16x16x32_f16(
      /*neg_a=*/false, a, /*neg_b=*/false, b,
      /*c_mod=*/(short)0, c, /*reuse_a=*/false, /*reuse_b=*/false);
}

static __device__ __forceinline__ v16h cat8(v8h lo, v8h hi) {
  v16h r;
#pragma unroll
  for (int i = 0; i < 8; ++i) { r[i] = lo[i]; r[i + 8] = hi[i]; }
  return r;
}

static __device__ __forceinline__ v8f v8f_zero() {
  v8f z;
#pragma unroll
  for (int i = 0; i < 8; ++i) z[i] = 0.0f;
  return z;
}

// ---------------------------------------------------------------------------
// TDM: async 2D tile load Global -> LDS (D# groups 0+1 per CDNA5 ISA 8.3/8.4).
// 2-group form: tensors up to 2D, VADDR2/VADDR3 = NULL. Tracked by TENSORcnt.
// All operands wave-uniform -> land in SGPRs for the "s" constraints.
// ---------------------------------------------------------------------------
static __device__ __forceinline__ void tdm_load_2d(
    unsigned int lds_addr, const void* gaddr,
    unsigned int tensor_d0, unsigned int tensor_d1,
    unsigned int tile_d0, unsigned int tile_d1,
    unsigned long long stride0_elems) {
  unsigned long long ga = (unsigned long long)(uintptr_t)gaddr;
  u32x4 g0;
  g0[0] = 1u;                                     // count=1, user descriptor
  g0[1] = lds_addr;                               // lds_addr [63:32]
  g0[2] = (unsigned int)ga;                       // global_addr [95:64]
  g0[3] = (unsigned int)((ga >> 32) & 0x1FFFFFFu) // global_addr [120:96]
          | 0x80000000u;                          // type=2 ("image") [127:126]
  u32x8 g1;
  g1[0] = (1u << 16);                             // data_size=1 (2B), wg_mask=0
  g1[1] = (tensor_d0 & 0xFFFFu) << 16;            // tensor_dim0[15:0]  @bits 63:48
  g1[2] = (tensor_d0 >> 16) | ((tensor_d1 & 0xFFFFu) << 16);
  g1[3] = (tensor_d1 >> 16) | (tile_d0 << 16);    // tile_dim0 @bits 127:112
  g1[4] = (tile_d1 & 0xFFFFu);                    // tile_dim1; tile_dim2=0
  g1[5] = (unsigned int)stride0_elems;            // tensor_dim0_stride[31:0]
  g1[6] = (unsigned int)(stride0_elems >> 32) & 0xFFFFu;
  g1[7] = 0u;
  asm volatile("tensor_load_to_lds %0, %1" :: "s"(g0), "s"(g1) : "memory");
}

// ---------------------------------------------------------------------------
// Convert f32 weight (256x256, W[k][n]) -> f16 transposed Wt[n][k].
// ---------------------------------------------------------------------------
__global__ void wt_cvt_transpose(const float* __restrict__ w, h16* __restrict__ wt) {
  int i = blockIdx.x * blockDim.x + threadIdx.x;   // 65536 threads
  int k = i >> 8;
  int n = i & 255;
  wt[n * HIDDEN + k] = (h16)w[k * HIDDEN + n];
}

// ---------------------------------------------------------------------------
// Projection: Y = X @ W + b (X f32 nrows x 256, Wt f16 transposed)
// mode 0: Q -> RoPE(cos[row]) * 1/sqrt(256), row-major f16 out
// mode 1: K -> RoPE(cos[row/rf]) for row < limit = SK - nex, row-major f16 out
// mode 2: V -> no RoPE, TRANSPOSED f16 out (out[col*SK_TOT + row])
// ---------------------------------------------------------------------------
__global__ __launch_bounds__(128) void proj_kernel(
    const float* __restrict__ x, const h16* __restrict__ wt,
    const float* __restrict__ bias,
    const float* __restrict__ cosp, const float* __restrict__ sinp,
    h16* __restrict__ out, int nrows, int mode, const int* __restrict__ nexp) {
  int wave = (blockIdx.x * blockDim.x + threadIdx.x) >> 5;
  int lane = threadIdx.x & 31;
  int ntiles = (nrows >> 4) * 16;
  if (wave >= ntiles) return;

  int rbase = (wave >> 4) * 16;
  int cbase = (wave & 15) * 16;
  int ln = lane & 15;
  int hi = lane >> 4;
  int col = cbase + ln;

  const float* xrow = x + (rbase + ln) * HIDDEN;   // A: M = lane%16
  const h16*   wrow = wt + col * HIDDEN;           // B: N = lane%16

  v8f acc = v8f_zero();
#pragma unroll
  for (int kc = 0; kc < 8; ++kc) {
    int kbA = kc * 32 + hi * 8;   // A: lane<16 K{0..7,16..23}, lane>=16 {8..15,24..31}
    v4f x0 = *(const v4f*)(xrow + kbA);
    v4f x1 = *(const v4f*)(xrow + kbA + 4);
    v4f x2 = *(const v4f*)(xrow + kbA + 16);
    v4f x3 = *(const v4f*)(xrow + kbA + 20);
    v16h a;
#pragma unroll
    for (int i = 0; i < 4; ++i) {
      a[i]      = (h16)x0[i];
      a[4 + i]  = (h16)x1[i];
      a[8 + i]  = (h16)x2[i];
      a[12 + i] = (h16)x3[i];
    }
    int kbB = kc * 32 + hi * 8;   // B: lane<16 K{0..7,16..23}, lane>=16 {8..15,24..31}
    v8h b0 = *(const v8h*)(wrow + kbB);
    v8h b1 = *(const v8h*)(wrow + kbB + 16);
    acc = wmma16(a, cat8(b0, b1), acc);
  }

  float bv = bias[col];
  float res[8];
#pragma unroll
  for (int r = 0; r < 8; ++r) res[r] = acc[r] + bv;

  if (mode <= 1) {
    int limit = (mode == 1) ? (SK_TOT - *nexp) : nrows;
    int rf = limit / SQ; if (rf < 1) rf = 1;
#pragma unroll
    for (int r = 0; r < 8; ++r) {
      int row = rbase + r + hi * 8;
      float v  = res[r];
      float pv = __shfl_xor(v, 1, 32);          // RoPE pair partner (col ^ 1)
      if (mode == 0 || row < limit) {
        int crow = (mode == 0) ? row : (row / rf);
        float cv = cosp[crow * HIDDEN + col];
        float sv = sinp[crow * HIDDEN + col];
        float rot = (col & 1) ? pv : -pv;       // rotate_pairwise: (x0,x1)->(-x1,x0)
        v = v * cv + rot * sv;
      }
      if (mode == 0) v *= 0.0625f;              // fold 1/sqrt(256) into Q
      out[row * HIDDEN + col] = (h16)v;
    }
  } else {
    v8h pk;
#pragma unroll
    for (int r = 0; r < 8; ++r) pk[r] = (h16)res[r];
    *(v8h*)(out + col * SK_TOT + rbase + hi * 8) = pk;
  }
}

// ---------------------------------------------------------------------------
// Flash attention, cooperative block: 4 waves x 16 query rows = 64 rows/block.
// K/V tiles (32 keys) staged in LDS by the Tensor Data Mover, double-buffered:
//   kbuf[i] = smem + i*8192       (32 x 256 f16, 16KB each)
//   vbuf[i] = smem + 16384+i*8192 (256 x 32 f16, 16KB each)
// Wave 0 issues tensor_load_to_lds for tile t+1 while all waves compute on t.
// P staging reuses the first 4KB of the current (dead) K buffer.
// ---------------------------------------------------------------------------
__global__ __launch_bounds__(128) void attn_kernel(
    const h16* __restrict__ qe, const h16* __restrict__ ke,
    const h16* __restrict__ vt, h16* __restrict__ o16) {
  extern __shared__ h16 smem[];                  // 64KB dynamic

  int wid  = threadIdx.x >> 5;
  int lane = threadIdx.x & 31;
  int qbase = (blockIdx.x * 4 + wid) * 16;
  int ln = lane & 15;
  int hi = lane >> 4;

  // Preload Q A-fragments for all 8 K-chunks of d=256
  v16h qf[8];
  const h16* qrow = qe + (qbase + ln) * HIDDEN;
#pragma unroll
  for (int kc = 0; kc < 8; ++kc) {
    int kb = kc * 32 + hi * 8;
    qf[kc] = cat8(*(const v8h*)(qrow + kb), *(const v8h*)(qrow + kb + 16));
  }

  float m[8], lsum[8];
  v8f o[16];
#pragma unroll
  for (int r = 0; r < 8; ++r) { m[r] = -1.0e30f; lsum[r] = 0.0f; }
#pragma unroll
  for (int c = 0; c < 16; ++c) o[c] = v8f_zero();

  // Prologue: DMA tile 0 into buffer 0
  if (wid == 0) {
    tdm_load_2d((unsigned int)(uintptr_t)(smem), ke,
                HIDDEN, SK_TOT, HIDDEN, KT, HIDDEN);
    tdm_load_2d((unsigned int)(uintptr_t)(smem + 16384), vt,
                SK_TOT, HIDDEN, KT, HIDDEN, SK_TOT);
    __builtin_amdgcn_s_wait_tensorcnt(0);
  }
  __syncthreads();

  int cur = 0;
  for (int t = 0; t < NKT; ++t) {
    h16* kcur = smem + (cur << 13);              // current K tile (32 x 256)
    h16* vcur = smem + 16384 + (cur << 13);      // current V^T tile (256 x 32)
    h16* knxt = smem + ((cur ^ 1) << 13);
    h16* vnxt = smem + 16384 + ((cur ^ 1) << 13);

    // Issue DMA for tile t+1 into the other buffer (overlaps with compute)
    if (wid == 0 && t + 1 < NKT) {
      int kt1 = (t + 1) * KT;
      tdm_load_2d((unsigned int)(uintptr_t)knxt, ke + (size_t)kt1 * HIDDEN,
                  HIDDEN, SK_TOT, HIDDEN, KT, HIDDEN);
      tdm_load_2d((unsigned int)(uintptr_t)vnxt, vt + kt1,
                  SK_TOT, HIDDEN, KT, HIDDEN, SK_TOT);
    }

    // ---- scores: S = Q * K^T from the LDS K tile ----
    v8f sa = v8f_zero(), sb = v8f_zero();
    const h16* krowA = kcur + ln * HIDDEN;              // B: N = key index
    const h16* krowB = krowA + 16 * HIDDEN;
#pragma unroll
    for (int kc = 0; kc < 8; ++kc) {
      int kb = kc * 32 + hi * 8;
      sa = wmma16(qf[kc], cat8(*(const v8h*)(krowA + kb),
                               *(const v8h*)(krowA + kb + 16)), sa);
      sb = wmma16(qf[kc], cat8(*(const v8h*)(krowB + kb),
                               *(const v8h*)(krowB + kb + 16)), sb);
    }
    __syncthreads();   // all waves done reading K tile -> its LDS is reusable

    // ---- online softmax (row = r + 8*hi, 16 cols in a 16-lane half) ----
    float pa[8], pb[8], alpha[8];
#pragma unroll
    for (int r = 0; r < 8; ++r) {
      float tm = fmaxf(sa[r], sb[r]);
#pragma unroll
      for (int d = 1; d < 16; d <<= 1) tm = fmaxf(tm, __shfl_xor(tm, d, 16));
      float mn = fmaxf(m[r], tm);
      alpha[r] = __expf(m[r] - mn);
      m[r] = mn;
      pa[r] = __expf(sa[r] - mn);
      pb[r] = __expf(sb[r] - mn);
      float rs = pa[r] + pb[r];
#pragma unroll
      for (int d = 1; d < 16; d <<= 1) rs += __shfl_xor(rs, d, 16);
      lsum[r] = lsum[r] * alpha[r] + rs;
    }
#pragma unroll
    for (int c = 0; c < 16; ++c)
#pragma unroll
      for (int r = 0; r < 8; ++r) o[c][r] *= alpha[r];

    // ---- stage P (16x32 f16) C-layout -> A-layout via dead K-buffer LDS ----
    h16* myp = kcur + wid * (16 * KT);
#pragma unroll
    for (int r = 0; r < 8; ++r) {
      int row = r + hi * 8;
      myp[row * KT + ln]      = (h16)pa[r];
      myp[row * KT + 16 + ln] = (h16)pb[r];
    }
    asm volatile("" ::: "memory");   // keep ds stores before ds loads (LDS in-order)
    int pb0 = ln * KT + hi * 8;
    v16h pf = cat8(*(const v8h*)(myp + pb0), *(const v8h*)(myp + pb0 + 16));
    asm volatile("" ::: "memory");

    // ---- O += P * V from the LDS V^T tile (rows are d-cols) ----
#pragma unroll
    for (int c = 0; c < 16; ++c) {
      const h16* vrow = vcur + (c * 16 + ln) * KT + hi * 8;
      o[c] = wmma16(pf, cat8(*(const v8h*)(vrow), *(const v8h*)(vrow + 16)), o[c]);
    }

    if (wid == 0) __builtin_amdgcn_s_wait_tensorcnt(0);  // tile t+1 landed
    __syncthreads();   // everyone done with buffers; next tile visible
    cur ^= 1;
  }

  // ---- normalize and write f16 O ----
#pragma unroll
  for (int r = 0; r < 8; ++r) lsum[r] = 1.0f / lsum[r];
#pragma unroll
  for (int c = 0; c < 16; ++c) {
    int col = c * 16 + ln;
#pragma unroll
    for (int r = 0; r < 8; ++r)
      o16[(qbase + r + hi * 8) * HIDDEN + col] = (h16)(o[c][r] * lsum[r]);
  }
}

// ---------------------------------------------------------------------------
// Output projection: d_out(f32) = O16 @ o_w + o_b
// ---------------------------------------------------------------------------
__global__ __launch_bounds__(128) void proj_o_kernel(
    const h16* __restrict__ xin, const h16* __restrict__ wt,
    const float* __restrict__ bias, float* __restrict__ out) {
  int wave = (blockIdx.x * blockDim.x + threadIdx.x) >> 5;
  int lane = threadIdx.x & 31;
  int rbase = (wave >> 4) * 16;
  int cbase = (wave & 15) * 16;
  int ln = lane & 15;
  int hi = lane >> 4;
  int col = cbase + ln;

  const h16* xrow = xin + (rbase + ln) * HIDDEN;
  const h16* wrow = wt + col * HIDDEN;
  v8f acc = v8f_zero();
#pragma unroll
  for (int kc = 0; kc < 8; ++kc) {
    int kbA = kc * 32 + hi * 8;
    v16h a = cat8(*(const v8h*)(xrow + kbA), *(const v8h*)(xrow + kbA + 16));
    int kbB = kc * 32 + hi * 8;
    v16h b = cat8(*(const v8h*)(wrow + kbB), *(const v8h*)(wrow + kbB + 16));
    acc = wmma16(a, b, acc);
  }
  float bv = bias[col];
#pragma unroll
  for (int r = 0; r < 8; ++r)
    out[(rbase + r + hi * 8) * HIDDEN + col] = acc[r] + bv;
}

// ---------------------------------------------------------------------------
extern "C" void kernel_launch(void* const* d_in, const int* in_sizes, int n_in,
                              void* d_out, int out_size, void* d_ws, size_t ws_size,
                              hipStream_t stream) {
  const float* query = (const float*)d_in[0];
  const float* key   = (const float*)d_in[1];
  const float* value = (const float*)d_in[2];
  const float* cosp  = (const float*)d_in[3];
  const float* sinp  = (const float*)d_in[4];
  const float* q_w   = (const float*)d_in[5];
  const float* q_b   = (const float*)d_in[6];
  const float* k_w   = (const float*)d_in[7];
  const float* k_b   = (const float*)d_in[8];
  const float* v_w   = (const float*)d_in[9];
  const float* v_b   = (const float*)d_in[10];
  const float* o_w   = (const float*)d_in[11];
  const float* o_b   = (const float*)d_in[12];
  const int*   nexp  = (const int*)d_in[13];

  // workspace layout (f16 elements)
  h16* qwt = (h16*)d_ws;                       // 4 x 256*256 transposed f16 weights
  h16* kwt = qwt + HIDDEN * HIDDEN;
  h16* vwt = kwt + HIDDEN * HIDDEN;
  h16* owt = vwt + HIDDEN * HIDDEN;
  h16* qe  = owt + HIDDEN * HIDDEN;            // 4096 x 256
  h16* ke  = qe  + (size_t)SQ * HIDDEN;        // 16448 x 256
  h16* vtm = ke  + (size_t)SK_TOT * HIDDEN;    // 256 x 16448 (V^T)
  h16* o16 = vtm + (size_t)SK_TOT * HIDDEN;    // 4096 x 256

  wt_cvt_transpose<<<256, 256, 0, stream>>>(q_w, qwt);
  wt_cvt_transpose<<<256, 256, 0, stream>>>(k_w, kwt);
  wt_cvt_transpose<<<256, 256, 0, stream>>>(v_w, vwt);
  wt_cvt_transpose<<<256, 256, 0, stream>>>(o_w, owt);

  proj_kernel<<<1024, 128, 0, stream>>>(query, qwt, q_b, cosp, sinp, qe, SQ, 0, nexp);
  proj_kernel<<<4112, 128, 0, stream>>>(key,   kwt, k_b, cosp, sinp, ke,  SK_TOT, 1, nexp);
  proj_kernel<<<4112, 128, 0, stream>>>(value, vwt, v_b, cosp, sinp, vtm, SK_TOT, 2, nexp);

  // attention: 64 blocks x (4 waves x 16 query rows), 64KB dynamic LDS
  attn_kernel<<<64, 128, 65536, stream>>>(qe, ke, vtm, o16);

  proj_o_kernel<<<1024, 128, 0, stream>>>(o16, owt, o_b, (float*)d_out);
}
